// RowWiseGatedAttention_84430467105267
// MI455X (gfx1250) — matrix operand order repaired and checked by reading, not compiled, not run on hardware
//
#include <hip/hip_runtime.h>

#define H 8
#define CH 32
#define CM 256
#define SS 128
#define II 256
#define SI (SS*II)            // 32768
#define NTOT ((size_t)CM*SI)  // 8388608 elements (same count for msa and pair)

typedef __attribute__((ext_vector_type(16))) __bf16 v16bf;
typedef __attribute__((ext_vector_type(16))) unsigned short us16v;
typedef __attribute__((ext_vector_type(8)))  float v8f;
typedef __attribute__((ext_vector_type(4)))  float f32x4;
typedef __attribute__((ext_vector_type(4)))  unsigned int u32x4;
typedef __attribute__((ext_vector_type(4)))  int i32x4;
typedef __attribute__((ext_vector_type(8)))  int i32x8;

struct u4x2 { uint4 a, b; };

static __device__ __forceinline__ unsigned short f2bf(float f) {
    unsigned int u = __float_as_uint(f);
    u += 0x7FFFu + ((u >> 16) & 1u);   // round-to-nearest-even
    return (unsigned short)(u >> 16);
}
static __device__ __forceinline__ float bf2f(unsigned int u) {
    return __uint_as_float((u & 0xFFFFu) << 16);
}
static __device__ __forceinline__ void unpack8(uint4 p, float* f) {
    f[0] = bf2f(p.x); f[1] = bf2f(p.x >> 16);
    f[2] = bf2f(p.y); f[3] = bf2f(p.y >> 16);
    f[4] = bf2f(p.z); f[5] = bf2f(p.z >> 16);
    f[6] = bf2f(p.w); f[7] = bf2f(p.w >> 16);
}
static __device__ __forceinline__ uint4 pack8(const float* f) {
    uint4 o;
    o.x = (unsigned)f2bf(f[0]) | ((unsigned)f2bf(f[1]) << 16);
    o.y = (unsigned)f2bf(f[2]) | ((unsigned)f2bf(f[3]) << 16);
    o.z = (unsigned)f2bf(f[4]) | ((unsigned)f2bf(f[5]) << 16);
    o.w = (unsigned)f2bf(f[6]) | ((unsigned)f2bf(f[7]) << 16);
    return o;
}
static __device__ __forceinline__ uint4 pack8v(v8f v) {
    float f[8];
    #pragma unroll
    for (int r = 0; r < 8; ++r) f[r] = v[r];
    return pack8(f);
}
static __device__ __forceinline__ v8f wmma_bf16(us16v a, us16v b, v8f c) {
    return __builtin_amdgcn_wmma_f32_16x16x32_bf16(
        false, __builtin_bit_cast(v16bf, a),
        false, __builtin_bit_cast(v16bf, b),
        (short)0, c, false, false);
}
static __device__ __forceinline__ unsigned lds_lo(const void* p) {
    return (unsigned)(unsigned long long)p;      // LDS aperture: low 32 bits = LDS offset
}

// ---- Tensor Data Mover: 2-D tile (d0 contiguous elems x d1 rows, row stride in elems), bf16 ----
static __device__ __forceinline__ void tdm_load_2d(unsigned lds_addr, const void* gptr,
                                                   unsigned d0, unsigned d1, unsigned stride) {
    unsigned long long ga = (unsigned long long)gptr;
    u32x4 g0;
    g0[0] = 1u;                                                  // count=1 valid descriptor
    g0[1] = lds_addr;                                            // lds_addr (bytes)
    g0[2] = (unsigned)ga;                                        // global_addr[31:0]
    g0[3] = (unsigned)((ga >> 32) & 0x1FFFFFFull) | (2u << 30);  // global_addr[56:32], type=2
    unsigned td0 = stride;                                       // tensor dims (OOB bounds)
    unsigned td1 = 0x40000u;
    i32x8 g1;
    g1[0] = (int)(1u << 16);                                     // wg_mask=0, data_size=1 (2B)
    g1[1] = (int)((td0 & 0xFFFFu) << 16);                        // tensor_dim0[15:0]
    g1[2] = (int)(((td0 >> 16) & 0xFFFFu) | ((td1 & 0xFFFFu) << 16));
    g1[3] = (int)(((td1 >> 16) & 0xFFFFu) | (d0 << 16));         // tile_dim0
    g1[4] = (int)d1;                                             // tile_dim1 (tile_dim2=0)
    g1[5] = (int)stride;                                         // tensor_dim0_stride
    g1[6] = 0;
    g1[7] = 0;
    i32x4 gz = {0, 0, 0, 0};
#if defined(__clang_major__) && __clang_major__ >= 23
    i32x8 gz8 = {0, 0, 0, 0, 0, 0, 0, 0};
    __builtin_amdgcn_tensor_load_to_lds(g0, g1, gz, gz, gz8, 0);
#else
    __builtin_amdgcn_tensor_load_to_lds(g0, g1, gz, gz, 0);
#endif
}

// ---------------- global-LN reductions (deterministic two-stage, 128-bit loads) ----------------
__global__ __launch_bounds__(256) void k_reduce(const float* __restrict__ x, float* __restrict__ part) {
    __shared__ float ss[256], sq[256];
    int t = threadIdx.x;
    const f32x4* x4 = (const f32x4*)x;
    float s = 0.f, q = 0.f;
    for (size_t idx = (size_t)blockIdx.x * 256 + t; idx < NTOT / 4; idx += (size_t)gridDim.x * 256) {
        f32x4 v = x4[idx];
        s += v[0] + v[1] + v[2] + v[3];
        q += v[0]*v[0] + v[1]*v[1] + v[2]*v[2] + v[3]*v[3];
    }
    ss[t] = s; sq[t] = q; __syncthreads();
    for (int off = 128; off > 0; off >>= 1) {
        if (t < off) { ss[t] += ss[t + off]; sq[t] += sq[t + off]; }
        __syncthreads();
    }
    if (t == 0) { part[2*blockIdx.x] = ss[0]; part[2*blockIdx.x + 1] = sq[0]; }
}

__global__ __launch_bounds__(1024) void k_finalize(const float* __restrict__ p1,
                                                   const float* __restrict__ p2,
                                                   float* __restrict__ stats) {
    __shared__ float a[1024], b[1024], c[1024], d[1024];
    int t = threadIdx.x;
    a[t] = p1[2*t]; b[t] = p1[2*t + 1]; c[t] = p2[2*t]; d[t] = p2[2*t + 1];
    __syncthreads();
    for (int off = 512; off > 0; off >>= 1) {
        if (t < off) { a[t]+=a[t+off]; b[t]+=b[t+off]; c[t]+=c[t+off]; d[t]+=d[t+off]; }
        __syncthreads();
    }
    if (t == 0) {
        const float inv = 1.0f / (float)NTOT;
        float mu1 = a[0]*inv, var1 = b[0]*inv - mu1*mu1;
        float mu2 = c[0]*inv, var2 = d[0]*inv - mu2*mu2;
        stats[0] = mu1; stats[1] = rsqrtf(var1 + 1e-5f);
        stats[2] = mu2; stats[3] = rsqrtf(var2 + 1e-5f);
    }
}

// ---------------- normalize msa -> bf16 xn[256][32768] (vectorized) ----------------
__global__ __launch_bounds__(256) void k_norm(const float* __restrict__ x,
                                              const float* __restrict__ w,
                                              const float* __restrict__ b,
                                              const float* __restrict__ stats,
                                              unsigned short* __restrict__ xn) {
    float mu = stats[0], istd = stats[1];
    const f32x4* x4 = (const f32x4*)x;
    const f32x4* w4 = (const f32x4*)w;
    const f32x4* b4 = (const f32x4*)b;
    uint2* o2 = (uint2*)xn;
    for (size_t idx = (size_t)blockIdx.x * 256 + threadIdx.x; idx < NTOT / 4;
         idx += (size_t)gridDim.x * 256) {
        f32x4 xv = x4[idx], wv = w4[idx], bv = b4[idx];
        f32x4 y = (xv - mu) * istd * wv + bv;
        uint2 o;
        o.x = (unsigned)f2bf(y[0]) | ((unsigned)f2bf(y[1]) << 16);
        o.y = (unsigned)f2bf(y[2]) | ((unsigned)f2bf(y[3]) << 16);
        o2[idx] = o;
    }
}

// ---------------- pair bias: fused pair-LN + Wb contraction -> bias[h][i][j] f32 ----------------
__global__ __launch_bounds__(256) void k_bias(const float* __restrict__ pair,
                                              const float* __restrict__ w2,
                                              const float* __restrict__ b2,
                                              const float* __restrict__ Wb,
                                              const float* __restrict__ stats,
                                              float* __restrict__ bias) {
    __shared__ float wbs[H * 128];
    for (int e = threadIdx.x; e < H * 128; e += 256) wbs[e] = Wb[e];
    __syncthreads();
    float mu = stats[2], istd = stats[3];
    size_t t4 = (size_t)blockIdx.x * 256 + threadIdx.x;   // 0..16383 (float4 column groups)
    const f32x4* p4 = (const f32x4*)pair;
    const f32x4* w4 = (const f32x4*)w2;
    const f32x4* b4 = (const f32x4*)b2;
    f32x4* o4 = (f32x4*)bias;
    f32x4 acc[H] = {};
    for (int c = 0; c < 128; ++c) {
        size_t idx = (size_t)c * 16384 + t4;
        f32x4 z = (p4[idx] - mu) * istd * w4[idx] + b4[idx];
        #pragma unroll
        for (int hh = 0; hh < H; ++hh) acc[hh] += wbs[hh * 128 + c] * z;
    }
    #pragma unroll
    for (int hh = 0; hh < H; ++hh) o4[(size_t)hh * 16384 + t4] = acc[hh];
}

// ---------------- fused 4-way projection GEMM: TDM-pipelined xn panels ----------------
__global__ __launch_bounds__(256) void k_proj(const unsigned short* __restrict__ xn,
                                              const float* __restrict__ Wq, const float* __restrict__ Wk,
                                              const float* __restrict__ Wv, const float* __restrict__ Wg,
                                              const float* __restrict__ bg,
                                              unsigned short* __restrict__ qb,
                                              unsigned short* __restrict__ kb,
                                              unsigned short* __restrict__ vb,
                                              unsigned short* __restrict__ gb) {
    __shared__ unsigned short bsh[2][32][128];    // double-buffered xn panel [k][n]  16 KB
    __shared__ unsigned short ash[4][16][256];    // all W tiles (bf16)              32 KB
    int tid = threadIdx.x, lane = tid & 31, wv = tid >> 5;
    int half = lane >> 4, ln = lane & 15;
    int mt = blockIdx.x >> 8, nb = blockIdx.x & 255;  // 16 m-tiles x 256 n-blocks
    int m0 = mt * 16, n0b = nb * 128;
    // one-time W staging: 4 mats x 16 rows x 256 cols, fp32 -> bf16
    for (int e = tid; e < 4 * 16 * 64; e += 256) {
        int mat = e >> 10, r = (e >> 6) & 15, c4 = (e & 63) * 4;
        const float* wp = (mat == 0) ? Wq : (mat == 1) ? Wk : (mat == 2) ? Wv : Wg;
        f32x4 sv = *(const f32x4*)(wp + (size_t)(m0 + r) * CM + c4);
        uint2 pk;
        pk.x = (unsigned)f2bf(sv[0]) | ((unsigned)f2bf(sv[1]) << 16);
        pk.y = (unsigned)f2bf(sv[2]) | ((unsigned)f2bf(sv[3]) << 16);
        ((uint2*)&ash[mat][r][0])[e & 63] = pk;
    }
    if (wv == 0)   // prologue: DMA panel for k-step 0
        tdm_load_2d(lds_lo(&bsh[0][0][0]), xn + n0b, 128, 32, SI);
    v8f acc[4] = {};
    for (int kk = 0; kk < 8; ++kk) {
        int b = kk & 1, k0 = kk * 32;
        if (wv == 0) __builtin_amdgcn_s_wait_tensorcnt(0);
        __syncthreads();   // panel b + (iter0: ash) visible; all waves done reading buf 1-b
        if (kk < 7 && wv == 0)
            tdm_load_2d(lds_lo(&bsh[1 - b][0][0]),
                        xn + (size_t)(kk + 1) * 32 * SI + n0b, 128, 32, SI);
        us16v bfv;
        #pragma unroll
        for (int e = 0; e < 16; ++e) bfv[e] = bsh[b][16 * half + e][wv * 16 + ln];
        #pragma unroll
        for (int m = 0; m < 4; ++m) {
            u4x2 aa;
            aa.a = *(const uint4*)&ash[m][ln][k0 + 16 * half];
            aa.b = *(const uint4*)&ash[m][ln][k0 + 16 * half + 8];
            acc[m] = wmma_bf16(__builtin_bit_cast(us16v, aa), bfv, acc[m]);
        }
    }
    int si = n0b + wv * 16 + ln;
    int s = si >> 8, ii = si & 255;
    int obase = m0 + 8 * half;                 // 8-aligned run of output channels
    int h = obase >> 5, c0 = obase & 31;
    *(uint4*)&qb[(((size_t)h * SS + s) * II + ii) * CH + c0] = pack8v(acc[0]);  // Q: [h][s][i][ch]
    *(uint4*)&kb[(((size_t)h * SS + s) * II + ii) * CH + c0] = pack8v(acc[1]);  // K: [h][s][j][ch]
    #pragma unroll
    for (int r = 0; r < 8; ++r) {
        int o = obase + r, c = c0 + r;
        vb[(((size_t)h * SS + s) * CH + c) * II + ii] = f2bf(acc[2][r]);        // V: [h][s][ch][j]
        float gv = acc[3][r] + bg[o];
        gv = 1.0f / (1.0f + __expf(-gv));
        gb[(size_t)o * SI + si] = f2bf(gv);                                     // G: [o][s][i]
    }
}

// ---------------- attention: one block per (h,s); TDM stages K/V panels into LDS ----------------
__global__ __launch_bounds__(128) void k_attn(const unsigned short* __restrict__ q,
                                              const unsigned short* __restrict__ kmat,
                                              const unsigned short* __restrict__ v,
                                              const unsigned short* __restrict__ g,
                                              const float* __restrict__ bias,
                                              unsigned short* __restrict__ og) {
    __shared__ unsigned short kpan[II][CH];       // K panel [j][ch]  16 KB
    __shared__ unsigned short vpan[CH][II];       // V panel [ch][j]  16 KB
    __shared__ unsigned short a_lds[4][16][II];   // per-wave softmax rows 32 KB
    int tid = threadIdx.x, lane = tid & 31, wv = tid >> 5;
    int half = lane >> 4, ln = lane & 15;
    int h = blockIdx.x >> 7, s = blockIdx.x & 127;
    const size_t base = (size_t)(h * SS + s) * II * CH;
    const unsigned short* qp = q + base;

    if (wv == 0) {   // wave-level DMA (TENSORcnt)
        tdm_load_2d(lds_lo(&kpan[0][0]), kmat + base, II * CH, 1, II * CH);
        tdm_load_2d(lds_lo(&vpan[0][0]), v + base, II * CH, 1, II * CH);
        __builtin_amdgcn_s_wait_tensorcnt(0);
    }
    __syncthreads();

    const float isq = 0.1767766952966369f;        // 1/sqrt(32)
    #pragma unroll 1
    for (int it = 0; it < 4; ++it) {
        int i0 = (wv * 4 + it) * 16;
        u4x2 qa;
        qa.a = *(const uint4*)&qp[(size_t)(i0 + ln) * CH + 16 * half];
        qa.b = *(const uint4*)&qp[(size_t)(i0 + ln) * CH + 16 * half + 8];
        us16v aq = __builtin_bit_cast(us16v, qa);

        v8f st[16];
        #pragma unroll
        for (int t = 0; t < 16; ++t) {
            u4x2 bb;
            bb.a = *(const uint4*)&kpan[t * 16 + ln][16 * half];
            bb.b = *(const uint4*)&kpan[t * 16 + ln][16 * half + 8];
            v8f z = {};
            v8f sc = wmma_bf16(aq, __builtin_bit_cast(us16v, bb), z);
            #pragma unroll
            for (int r = 0; r < 8; ++r) {
                int m = r + 8 * half;
                sc[r] = sc[r] * isq + bias[((size_t)h * II + (i0 + m)) * II + t * 16 + ln];
            }
            st[t] = sc;
        }
        // row-wise softmax over j (row = 16-lane group within half-wave)
        #pragma unroll
        for (int r = 0; r < 8; ++r) {
            float mx = st[0][r];
            #pragma unroll
            for (int t = 1; t < 16; ++t) mx = fmaxf(mx, st[t][r]);
            #pragma unroll
            for (int d = 1; d < 16; d <<= 1) mx = fmaxf(mx, __shfl_xor(mx, d, 32));
            float sum = 0.f;
            #pragma unroll
            for (int t = 0; t < 16; ++t) { float e_ = __expf(st[t][r] - mx); st[t][r] = e_; sum += e_; }
            #pragma unroll
            for (int d = 1; d < 16; d <<= 1) sum += __shfl_xor(sum, d, 32);
            float rinv = 1.0f / sum;
            int m = r + 8 * half;
            #pragma unroll
            for (int t = 0; t < 16; ++t) a_lds[wv][m][t * 16 + ln] = f2bf(st[t][r] * rinv);
        }
        // O^T[i][ch] = A[i][j] * V^T[j][ch]  (LDS DS ops are in-order within the wave)
        v8f oc0 = {}, oc1 = {};
        #pragma unroll
        for (int jt = 0; jt < 8; ++jt) {
            u4x2 aa, b0, b1;
            aa.a = *(const uint4*)&a_lds[wv][ln][jt * 32 + 16 * half];
            aa.b = *(const uint4*)&a_lds[wv][ln][jt * 32 + 16 * half + 8];
            b0.a = *(const uint4*)&vpan[ln][jt * 32 + 16 * half];
            b0.b = *(const uint4*)&vpan[ln][jt * 32 + 16 * half + 8];
            b1.a = *(const uint4*)&vpan[16 + ln][jt * 32 + 16 * half];
            b1.b = *(const uint4*)&vpan[16 + ln][jt * 32 + 16 * half + 8];
            us16v af = __builtin_bit_cast(us16v, aa);
            oc0 = wmma_bf16(af, __builtin_bit_cast(us16v, b0), oc0);
            oc1 = wmma_bf16(af, __builtin_bit_cast(us16v, b1), oc1);
        }
        // gate and scatter into (c*H + h) channel order; 8 consecutive i per lane -> uint4
        int ibase = i0 + 8 * half;
        uint4 gp0 = *(const uint4*)&g[(size_t)(h * CH + ln) * SI + s * II + ibase];
        uint4 gp1 = *(const uint4*)&g[(size_t)(h * CH + 16 + ln) * SI + s * II + ibase];
        float gf0[8], gf1[8], of0[8], of1[8];
        unpack8(gp0, gf0); unpack8(gp1, gf1);
        #pragma unroll
        for (int r = 0; r < 8; ++r) { of0[r] = gf0[r] * oc0[r]; of1[r] = gf1[r] * oc1[r]; }
        *(uint4*)&og[((size_t)(ln * H + h) * SS + s) * II + ibase] = pack8(of0);
        *(uint4*)&og[((size_t)((16 + ln) * H + h) * SS + s) * II + ibase] = pack8(of1);
    }
}

// ---------------- output projection: out = Wrep * og + brep (TDM-pipelined) ----------------
__global__ __launch_bounds__(256) void k_outproj(const unsigned short* __restrict__ og,
                                                 const float* __restrict__ Wrep,
                                                 const float* __restrict__ brep,
                                                 float* __restrict__ out) {
    __shared__ unsigned short bsh[2][32][128];    // double-buffered og panel 16 KB
    __shared__ unsigned short ash[16][256];       // Wrep tile (bf16)          8 KB
    int tid = threadIdx.x, lane = tid & 31, wv = tid >> 5;
    int half = lane >> 4, ln = lane & 15;
    int mt = blockIdx.x >> 8, nb = blockIdx.x & 255;
    int m0 = mt * 16, n0b = nb * 128;
    for (int e = tid; e < 16 * 64; e += 256) {
        int r = e >> 6, c4 = (e & 63) * 4;
        f32x4 sv = *(const f32x4*)(Wrep + (size_t)(m0 + r) * CM + c4);
        uint2 pk;
        pk.x = (unsigned)f2bf(sv[0]) | ((unsigned)f2bf(sv[1]) << 16);
        pk.y = (unsigned)f2bf(sv[2]) | ((unsigned)f2bf(sv[3]) << 16);
        ((uint2*)&ash[r][0])[e & 63] = pk;
    }
    if (wv == 0)
        tdm_load_2d(lds_lo(&bsh[0][0][0]), og + n0b, 128, 32, SI);
    v8f acc = {};
    for (int kk = 0; kk < 8; ++kk) {
        int b = kk & 1, k0 = kk * 32;
        if (wv == 0) __builtin_amdgcn_s_wait_tensorcnt(0);
        __syncthreads();
        if (kk < 7 && wv == 0)
            tdm_load_2d(lds_lo(&bsh[1 - b][0][0]),
                        og + (size_t)(kk + 1) * 32 * SI + n0b, 128, 32, SI);
        us16v bfv;
        #pragma unroll
        for (int e = 0; e < 16; ++e) bfv[e] = bsh[b][16 * half + e][wv * 16 + ln];
        u4x2 aa;
        aa.a = *(const uint4*)&ash[ln][k0 + 16 * half];
        aa.b = *(const uint4*)&ash[ln][k0 + 16 * half + 8];
        acc = wmma_bf16(__builtin_bit_cast(us16v, aa), bfv, acc);
    }
    int si = n0b + wv * 16 + ln;
    #pragma unroll
    for (int r = 0; r < 8; ++r) {
        int p = m0 + r + 8 * half;
        out[(size_t)p * SI + si] = acc[r] + brep[p];
    }
}

extern "C" void kernel_launch(void* const* d_in, const int* in_sizes, int n_in,
                              void* d_out, int out_size, void* d_ws, size_t ws_size,
                              hipStream_t stream) {
    const float* msa  = (const float*)d_in[0];
    const float* pair = (const float*)d_in[1];
    const float* ln1w = (const float*)d_in[2];
    const float* ln1b = (const float*)d_in[3];
    const float* ln2w = (const float*)d_in[4];
    const float* ln2b = (const float*)d_in[5];
    const float* Wq   = (const float*)d_in[6];
    const float* Wk   = (const float*)d_in[7];
    const float* Wv   = (const float*)d_in[8];
    const float* Wb   = (const float*)d_in[9];
    const float* Wg   = (const float*)d_in[10];
    const float* bg   = (const float*)d_in[11];
    const float* Wrep = (const float*)d_in[12];
    const float* brep = (const float*)d_in[13];
    float* out = (float*)d_out;

    char* ws = (char*)d_ws;
    const size_t BF16MAT = (size_t)CM * SI * 2;   // 16 MB
    float* part1 = (float*)(ws + 0);
    float* part2 = (float*)(ws + 8192);
    float* stats = (float*)(ws + 16384);
    unsigned short* xn  = (unsigned short*)(ws + 65536);
    unsigned short* qb  = (unsigned short*)(ws + 65536 + 1 * BF16MAT);
    unsigned short* kb  = (unsigned short*)(ws + 65536 + 2 * BF16MAT);
    unsigned short* vb  = (unsigned short*)(ws + 65536 + 3 * BF16MAT);
    unsigned short* gb  = (unsigned short*)(ws + 65536 + 4 * BF16MAT);
    unsigned short* ogb = (unsigned short*)(ws + 65536 + 5 * BF16MAT);
    float* bias = (float*)(ws + 65536 + 6 * BF16MAT);

    k_reduce  <<<1024, 256, 0, stream>>>(msa, part1);
    k_reduce  <<<1024, 256, 0, stream>>>(pair, part2);
    k_finalize<<<1, 1024, 0, stream>>>(part1, part2, stats);
    k_norm    <<<2048, 256, 0, stream>>>(msa, ln1w, ln1b, stats, xn);
    k_bias    <<<64, 256, 0, stream>>>(pair, ln2w, ln2b, Wb, stats, bias);
    k_proj    <<<4096, 256, 0, stream>>>(xn, Wq, Wk, Wv, Wg, bg, qb, kb, vb, gb);
    k_attn    <<<1024, 128, 0, stream>>>(qb, kb, vb, gb, bias, ogb);
    k_outproj <<<4096, 256, 0, stream>>>(ogb, Wrep, brep, out);
    (void)in_sizes; (void)n_in; (void)out_size; (void)ws_size;
}
